// TokenChoiceTopKRouter_34127810134129
// MI455X (gfx1250) — compile-verified, matrix operand order checked by reading, not run on hardware
//
#include <hip/hip_runtime.h>
#include <hip/hip_bf16.h>

// ---------------------------------------------------------------------------
// MoE token-choice router for MI455X (gfx1250, wave32, WMMA).
// Dominant cost: logits = x @ gate_w^T  (16384x7168 * 7168x256, ~60 GFLOP).
// Split fp32 -> bf16 (hi+lo); compute hi*hi + hi*lo + lo*hi with
// v_wmma_f32_16x16x32_bf16 for near-fp32 accuracy at ~2.7x the native f32
// WMMA rate. gate_w bf16 (7.3MB) is L2-resident; M-tile=64 + B-fragment
// sharing across M-subtiles keeps total L2 B-traffic ~3.7GB. LDS x-tile is
// double-buffered (one workgroup barrier per K-step).
// ---------------------------------------------------------------------------

typedef __attribute__((ext_vector_type(16))) __bf16 v16bf;
typedef __attribute__((ext_vector_type(8)))  __bf16 v8bf;
typedef __attribute__((ext_vector_type(8)))  float  v8f;

#define TOKENS   16384
#define DMODEL   7168
#define EXPERTS  256
#define NGROUPS  8
#define KEEPG    4
#define TOPK     8
#define MB       64      // token rows per workgroup
#define NB       128     // experts per workgroup (gridDim.y = 2)
#define KT       32      // K tile (matches WMMA bf16 K)

// ---------------------------------------------------------------------------
// Kernel 1: split gate_w fp32 -> bf16 hi/lo pair
// ---------------------------------------------------------------------------
__global__ __launch_bounds__(256) void split_weights(const float* __restrict__ w,
                                                     __bf16* __restrict__ whi,
                                                     __bf16* __restrict__ wlo,
                                                     int n) {
    int i = blockIdx.x * 256 + threadIdx.x;
    if (i < n) {
        float f = w[i];
        __bf16 h = (__bf16)f;
        whi[i] = h;
        wlo[i] = (__bf16)(f - (float)h);
    }
}

// ---------------------------------------------------------------------------
// Kernel 2: zero the per-expert histogram region of d_out
// ---------------------------------------------------------------------------
__global__ void zero_hist(float* __restrict__ hist) {
    hist[threadIdx.x] = 0.0f;
}

// ---------------------------------------------------------------------------
// Helper: v16bf fragment from two contiguous 8-element runs.
// ISA 7.12.2 (16-bit 16x32 A): lane L holds M=L%16; elems 0..7 -> K base..
// base+7, elems 8..15 -> K base+16..base+23, base = (L/16)*8.
// ---------------------------------------------------------------------------
__device__ inline v16bf frag16(const __bf16* __restrict__ p0,
                               const __bf16* __restrict__ p1) {
    v8bf a = *(const v8bf*)p0;
    v8bf b = *(const v8bf*)p1;
    v16bf r;
#pragma unroll
    for (int i = 0; i < 8; ++i) { r[i] = a[i]; r[8 + i] = b[i]; }
    return r;
}

// ---------------------------------------------------------------------------
// Kernel 3: GEMM (split-bf16 WMMA) + fused sigmoid.
// Block = 256 threads = 8 waves: wave = (mw in 0..1) x (nw in 0..3).
// Each wave: 2 M-subtiles x 2 N-subtiles (32 rows x 32 experts of C);
// B fragments are reused across the 2 M-subtiles.
// ---------------------------------------------------------------------------
__global__ __launch_bounds__(256) void gate_gemm_sigmoid(
    const float*  __restrict__ x,
    const __bf16* __restrict__ whi,
    const __bf16* __restrict__ wlo,
    float*        __restrict__ scores) {

    __shared__ __bf16 xs_hi[2][MB][KT];   // double-buffered x tile (hi)
    __shared__ __bf16 xs_lo[2][MB][KT];   // double-buffered x tile (lo)

    const int tid  = threadIdx.x;
    const int lane = tid & 31;
    const int wid  = tid >> 5;
    const int mw   = wid & 1;              // 32-row half of the 64-row tile
    const int nw   = wid >> 1;             // 32-expert slice of the 128
    const int row0 = blockIdx.x * MB;
    const int col0 = blockIdx.y * NB + nw * 32;

    // x staging: each thread loads two float4 (rows ldr and ldr+32)
    const int ldr = tid >> 3;              // 0..31
    const int ldc = (tid & 7) * 4;         // 0..28

    v8f acc[2][2];
#pragma unroll
    for (int i = 0; i < 2; ++i)
#pragma unroll
        for (int j = 0; j < 2; ++j)
            acc[i][j] = (v8f){0.f,0.f,0.f,0.f,0.f,0.f,0.f,0.f};

    const int base = (lane >> 4) * 8;      // K-run base per lane half

    // ---- prologue: stage tile 0 into buffer 0 ----
    {
        const float4 v0 = *(const float4*)(x + (size_t)(row0 + ldr)      * DMODEL + ldc);
        const float4 v1 = *(const float4*)(x + (size_t)(row0 + ldr + 32) * DMODEL + ldc);
        float f0[4] = {v0.x, v0.y, v0.z, v0.w};
        float f1[4] = {v1.x, v1.y, v1.z, v1.w};
#pragma unroll
        for (int j = 0; j < 4; ++j) {
            __bf16 h0 = (__bf16)f0[j];
            __bf16 h1 = (__bf16)f1[j];
            xs_hi[0][ldr][ldc + j]      = h0;
            xs_lo[0][ldr][ldc + j]      = (__bf16)(f0[j] - (float)h0);
            xs_hi[0][ldr + 32][ldc + j] = h1;
            xs_lo[0][ldr + 32][ldc + j] = (__bf16)(f1[j] - (float)h1);
        }
    }
    __syncthreads();

    for (int k0 = 0; k0 < DMODEL; k0 += KT) {
        const int cur = (k0 / KT) & 1;
        const int nxt = cur ^ 1;
        const bool has_next = (k0 + KT) < DMODEL;

        // ---- issue next x-tile global loads early (uniform branch) ----
        float4 nv0, nv1;
        if (has_next) {
            nv0 = *(const float4*)(x + (size_t)(row0 + ldr)      * DMODEL + k0 + KT + ldc);
            nv1 = *(const float4*)(x + (size_t)(row0 + ldr + 32) * DMODEL + k0 + KT + ldc);
        }

        // ---- A fragments for both M-subtiles (from LDS) ----
        v16bf a_hi[2], a_lo[2];
#pragma unroll
        for (int ms = 0; ms < 2; ++ms) {
            const int am = mw * 32 + ms * 16 + (lane & 15);
            a_hi[ms] = frag16(&xs_hi[cur][am][base], &xs_hi[cur][am][base + 16]);
            a_lo[ms] = frag16(&xs_lo[cur][am][base], &xs_lo[cur][am][base + 16]);
        }

        // ---- 2 N-subtiles; B fragments shared across the 2 M-subtiles ----
#pragma unroll
        for (int nt = 0; nt < 2; ++nt) {
            const int n = col0 + nt * 16 + (lane & 15);
            const __bf16* bh = whi + (size_t)n * DMODEL + k0 + base;
            const __bf16* bl = wlo + (size_t)n * DMODEL + k0 + base;
            v16bf b_hi = frag16(bh, bh + 16);
            v16bf b_lo = frag16(bl, bl + 16);
#pragma unroll
            for (int ms = 0; ms < 2; ++ms) {
                acc[ms][nt] = __builtin_amdgcn_wmma_f32_16x16x32_bf16(
                    false, a_hi[ms], false, b_hi, (short)0, acc[ms][nt], false, false);
                acc[ms][nt] = __builtin_amdgcn_wmma_f32_16x16x32_bf16(
                    false, a_hi[ms], false, b_lo, (short)0, acc[ms][nt], false, false);
                acc[ms][nt] = __builtin_amdgcn_wmma_f32_16x16x32_bf16(
                    false, a_lo[ms], false, b_hi, (short)0, acc[ms][nt], false, false);
            }
        }

        // ---- convert + store next tile into the other LDS buffer ----
        if (has_next) {
            float f0[4] = {nv0.x, nv0.y, nv0.z, nv0.w};
            float f1[4] = {nv1.x, nv1.y, nv1.z, nv1.w};
#pragma unroll
            for (int j = 0; j < 4; ++j) {
                __bf16 h0 = (__bf16)f0[j];
                __bf16 h1 = (__bf16)f1[j];
                xs_hi[nxt][ldr][ldc + j]      = h0;
                xs_lo[nxt][ldr][ldc + j]      = (__bf16)(f0[j] - (float)h0);
                xs_hi[nxt][ldr + 32][ldc + j] = h1;
                xs_lo[nxt][ldr + 32][ldc + j] = (__bf16)(f1[j] - (float)h1);
            }
        }
        __syncthreads();
    }

    // ---- epilogue: sigmoid via v_rcp_f32; C layout: VGPR r -> M=r / M=r+8 ----
#pragma unroll
    for (int ms = 0; ms < 2; ++ms)
#pragma unroll
        for (int nt = 0; nt < 2; ++nt) {
            const int n = col0 + nt * 16 + (lane & 15);
#pragma unroll
            for (int r = 0; r < 8; ++r) {
                const int m = row0 + mw * 32 + ms * 16 + (lane >> 4) * 8 + r;
                const float lg = acc[ms][nt][r];
                scores[(size_t)m * EXPERTS + n] =
                    __builtin_amdgcn_rcpf(1.0f + __expf(-lg));
            }
        }
}

// ---------------------------------------------------------------------------
// Kernel 4: group-limited top-k routing. One block (256 thr) per token.
// Wave g handles group g (32 experts == one wave32).
// ---------------------------------------------------------------------------
__global__ __launch_bounds__(256) void route_topk(
    const float* __restrict__ scores,
    const float* __restrict__ expert_bias,
    float* __restrict__ top_scores,   // (T, 8)
    float* __restrict__ sel_out,      // (T, 8) indices stored as float
    float* __restrict__ hist) {       // (256)

    const int t    = blockIdx.x;
    const int e    = threadIdx.x;
    const int lane = e & 31;
    const int wid  = e >> 5;
    const float NEGB = -3.4e38f;

    __shared__ float raw[EXPERTS];
    __shared__ float biased[EXPERTS];
    __shared__ float gscore[NGROUPS];
    __shared__ float wmax[NGROUPS];
    __shared__ int   widx[NGROUPS];
    __shared__ int   sel[TOPK];
    __shared__ int   keepmask_s;

    const float s = scores[(size_t)t * EXPERTS + e];
    raw[e] = s;
    float b = s + expert_bias[e];
    biased[e] = b;

    // ---- wave-level top-2 (group score = top1 + top2 of biased) ----
    float v1 = b, v2 = NEGB;
#pragma unroll
    for (int off = 16; off; off >>= 1) {
        float o1 = __shfl_down(v1, off, 32);
        float o2 = __shfl_down(v2, off, 32);
        if (o1 > v1) { v2 = fmaxf(v1, o2); v1 = o1; }
        else         { v2 = fmaxf(v2, o1); }
    }
    if (lane == 0) gscore[wid] = v1 + v2;
    __syncthreads();

    // ---- pick top-4 groups ----
    if (e == 0) {
        int mask = 0;
        for (int k = 0; k < KEEPG; ++k) {
            float best = NEGB; int bi = 0;
            for (int g = 0; g < NGROUPS; ++g)
                if (!((mask >> g) & 1) && gscore[g] > best) { best = gscore[g]; bi = g; }
            mask |= 1 << bi;
        }
        keepmask_s = mask;
    }
    __syncthreads();
    if (!((keepmask_s >> wid) & 1)) biased[e] = NEGB;
    __syncthreads();

    // ---- 8 rounds of masked argmax over 256 experts ----
    for (int k = 0; k < TOPK; ++k) {
        float v = biased[e]; int idx = e;
#pragma unroll
        for (int off = 16; off; off >>= 1) {
            float ov = __shfl_down(v, off, 32);
            int   oi = __shfl_down(idx, off, 32);
            if (ov > v) { v = ov; idx = oi; }
        }
        if (lane == 0) { wmax[wid] = v; widx[wid] = idx; }
        __syncthreads();
        if (e == 0) {
            float best = NEGB; int bi = 0;
            for (int g = 0; g < NGROUPS; ++g)
                if (wmax[g] > best) { best = wmax[g]; bi = widx[g]; }
            sel[k] = bi;
            biased[bi] = NEGB;
        }
        __syncthreads();
    }

    // ---- normalize unbiased scores, emit outputs + histogram ----
    if (e == 0) {
        float ts[TOPK];
        float sum = 1e-20f;
#pragma unroll
        for (int k = 0; k < TOPK; ++k) { ts[k] = raw[sel[k]]; sum += ts[k]; }
        const float inv = 2.5f / sum;
#pragma unroll
        for (int k = 0; k < TOPK; ++k) {
            top_scores[(size_t)t * TOPK + k] = ts[k] * inv;
            sel_out[(size_t)t * TOPK + k]    = (float)sel[k];
            atomicAdd(&hist[sel[k]], 1.0f);
        }
    }
}

// ---------------------------------------------------------------------------
extern "C" void kernel_launch(void* const* d_in, const int* in_sizes, int n_in,
                              void* d_out, int out_size, void* d_ws, size_t ws_size,
                              hipStream_t stream) {
    const float* x      = (const float*)d_in[0];   // (T, D)
    const float* gate_w = (const float*)d_in[1];   // (E, D)
    const float* bias   = (const float*)d_in[2];   // (E,)
    float* out = (float*)d_out;

    // workspace layout: w_hi | w_lo (bf16, E*D each) | scores (fp32, T*E)
    const size_t wcnt = (size_t)EXPERTS * DMODEL;
    char* ws = (char*)d_ws;
    __bf16* whi = (__bf16*)ws;
    __bf16* wlo = (__bf16*)(ws + wcnt * sizeof(__bf16));
    float* scores = (float*)(ws + 2 * wcnt * sizeof(__bf16));

    float* top_scores = out;
    float* sel_out    = out + (size_t)TOKENS * TOPK;
    float* hist       = out + 2 * (size_t)TOKENS * TOPK;

    split_weights<<<(int)((wcnt + 255) / 256), 256, 0, stream>>>(gate_w, whi, wlo, (int)wcnt);
    zero_hist<<<1, EXPERTS, 0, stream>>>(hist);
    dim3 ggrid(TOKENS / MB, EXPERTS / NB);
    gate_gemm_sigmoid<<<ggrid, 256, 0, stream>>>(x, whi, wlo, scores);
    route_topk<<<TOKENS, 256, 0, stream>>>(scores, bias, top_scores, sel_out, hist);
}
